// BigramLanguageModel_30193620091024
// MI455X (gfx1250) — matrix operand order repaired and checked by reading, not compiled, not run on hardware
//
#include <hip/hip_runtime.h>
#include <hip/hip_bf16.h>
#include <math.h>

#define EMBED  64
#define HEADS  8
#define DHEAD  8
#define LAYERS 4
#define VOCAB  32000
#define SEQ    1024
#define BATCH  8
#define BT     (BATCH * SEQ)   /* 8192 rows */
#define FF     (4 * EMBED)     /* 256 */
#define EPS    1e-5f

typedef __bf16 bf16_t;
typedef __attribute__((ext_vector_type(16))) __bf16 v16bf;
typedef __attribute__((ext_vector_type(8)))  __bf16 v8bf;
typedef __attribute__((ext_vector_type(4)))  __bf16 v4bf;
typedef __attribute__((ext_vector_type(8)))  float  v8f;

// ---------------------------------------------------------------- embedding
__global__ void k_embed(const int* __restrict__ x, const float* __restrict__ tok,
                        const float* __restrict__ pos, float* __restrict__ h) {
    int i = blockIdx.x * blockDim.x + threadIdx.x;   // over BT*EMBED
    if (i >= BT * EMBED) return;
    int bt = i / EMBED, c = i % EMBED;
    int t = bt % SEQ;
    h[i] = tok[x[bt] * EMBED + c] + pos[t * EMBED + c];
}

// ---------------------------------------------------------------- f32 -> bf16 (row-major, for A operands)
__global__ void k_f2b(const float* __restrict__ in, bf16_t* __restrict__ out, int n) {
    int i = blockIdx.x * blockDim.x + threadIdx.x;
    if (i < n) out[i] = (bf16_t)in[i];
}

// ---------------------------------------------------------------- f32 [K,N] -> packed bf16 B fragments
// Packed layout: [N/16][K/32][lane 0..31][16 halves], matching the wave32
// V_WMMA_*_16X16X32 B layout (lanes 0-15: N=lane, K=0..7,16..23; lanes 16-31: N=lane-16, K=8..15,24..31).
// After packing, each lane's fragment is one contiguous 32-byte load.
__global__ void k_pack_b(const float* __restrict__ in, bf16_t* __restrict__ out, int K, int N) {
    int idx = blockIdx.x * blockDim.x + threadIdx.x;
    if (idx >= K * N) return;
    int k = idx / N, n = idx % N;
    int k_t = k >> 5, kk = k & 31;
    int n_t = n >> 4, nn = n & 15;
    int lane = (((kk >> 3) & 1) << 4) + nn;
    int i = (kk & 7) | ((kk & 16) >> 1);
    size_t o = ((size_t)(n_t * (K >> 5) + k_t) * 32 + lane) * 16 + i;
    out[o] = (bf16_t)in[idx];
}

// ---------------------------------------------------------------- QKV projection
__global__ void k_qkv(const float* __restrict__ h,
                      const float* __restrict__ Wq, const float* __restrict__ bq,
                      const float* __restrict__ Wk, const float* __restrict__ bk,
                      const float* __restrict__ Wv, const float* __restrict__ bv,
                      float* __restrict__ q, float* __restrict__ k, float* __restrict__ v) {
    __shared__ float row[EMBED];
    int bt = blockIdx.x;
    int tid = threadIdx.x;
    if (tid < EMBED) row[tid] = h[bt * EMBED + tid];
    __syncthreads();
    int which = tid >> 6;          // 0=q, 1=k, 2=v
    int hd = tid & 63;
    int hh = hd / DHEAD, dd = hd % DHEAD;
    const float* W  = (which == 0) ? Wq : (which == 1) ? Wk : Wv;   // [H, EMBED, DHEAD]
    const float* bb = (which == 0) ? bq : (which == 1) ? bk : bv;   // [H, DHEAD]
    float acc = bb[hh * DHEAD + dd];
    const float* wp = W + hh * EMBED * DHEAD + dd;
#pragma unroll 8
    for (int c = 0; c < EMBED; ++c) acc += row[c] * wp[c * DHEAD];
    int b = bt / SEQ, t = bt % SEQ;
    float* dst = (which == 0) ? q : (which == 1) ? k : v;           // [B,H,T,D]
    dst[((b * HEADS + hh) * SEQ + t) * DHEAD + dd] = acc;
}

// ---------------------------------------------------------------- attention
// One block (256 threads) per (b,h,t) row. Faithful to reference:
//   kr[d,s] = k_flat[d*T + s]  (raw reshape, NOT a transpose)
//   w = tril(w); w==0 -> -inf; softmax(w / sqrt(DHEAD))
__global__ void k_attn(const float* __restrict__ q, const float* __restrict__ k,
                       const float* __restrict__ v, float* __restrict__ attn) {
    int idx = blockIdx.x;                 // bh*T + t
    int t  = idx % SEQ;
    int bh = idx / SEQ;
    int b  = bh / HEADS, hh = bh % HEADS;
    int tid = threadIdx.x;
    const float* qrow  = q + ((size_t)bh * SEQ + t) * DHEAD;
    const float* kbase = k + (size_t)bh * SEQ * DHEAD;   // flat [T*D], raw-reshaped to [D,T]
    const float* vbase = v + (size_t)bh * SEQ * DHEAD;

    float qv[DHEAD];
#pragma unroll
    for (int d = 0; d < DHEAD; ++d) qv[d] = qrow[d];

    const float scale = 0.35355339059327373f;   // 1/sqrt(8)
    float wv[4];
    int   ss[4];
#pragma unroll
    for (int j = 0; j < 4; ++j) {
        int s = tid + j * 256;
        ss[j] = s;
        float acc = 0.f;
#pragma unroll
        for (int d = 0; d < DHEAD; ++d) acc += qv[d] * kbase[d * SEQ + s];
        bool valid = (s <= t) && (acc != 0.0f);
        wv[j] = valid ? acc * scale : -INFINITY;
    }

    __shared__ float sm[256];
    float m = fmaxf(fmaxf(wv[0], wv[1]), fmaxf(wv[2], wv[3]));
    sm[tid] = m; __syncthreads();
    for (int st = 128; st > 0; st >>= 1) {
        if (tid < st) sm[tid] = fmaxf(sm[tid], sm[tid + st]);
        __syncthreads();
    }
    float rowmax = sm[0]; __syncthreads();

    float p[4], psum = 0.f;
#pragma unroll
    for (int j = 0; j < 4; ++j) {
        p[j] = (wv[j] == -INFINITY) ? 0.f : __expf(wv[j] - rowmax);
        psum += p[j];
    }
    sm[tid] = psum; __syncthreads();
    for (int st = 128; st > 0; st >>= 1) {
        if (tid < st) sm[tid] += sm[tid + st];
        __syncthreads();
    }
    float inv = 1.0f / sm[0]; __syncthreads();

    float oacc[DHEAD];
#pragma unroll
    for (int d = 0; d < DHEAD; ++d) oacc[d] = 0.f;
#pragma unroll
    for (int j = 0; j < 4; ++j) {
        float pj = p[j] * inv;
        const float* vrow = vbase + ss[j] * DHEAD;
#pragma unroll
        for (int d = 0; d < DHEAD; ++d) oacc[d] += pj * vrow[d];
    }

    __shared__ float osm[256 * DHEAD];
#pragma unroll
    for (int d = 0; d < DHEAD; ++d) osm[tid * DHEAD + d] = oacc[d];
    __syncthreads();
    for (int st = 128; st > 0; st >>= 1) {
        if (tid < st) {
#pragma unroll
            for (int d = 0; d < DHEAD; ++d)
                osm[tid * DHEAD + d] += osm[(tid + st) * DHEAD + d];
        }
        __syncthreads();
    }
    if (tid < DHEAD)   // o.transpose(0,2,1,3).reshape(B,T,C): head-concat layout
        attn[((size_t)(b * SEQ + t)) * EMBED + hh * DHEAD + tid] = osm[tid];
}

// ---------------------------------------------------------------- LN1 + residual:  h += LN(attn)*g + b
__global__ void k_ln1_res(const float* __restrict__ a, const float* __restrict__ g,
                          const float* __restrict__ bb, float* __restrict__ h) {
    __shared__ float s[EMBED];
    int bt = blockIdx.x, c = threadIdx.x;
    float x = a[bt * EMBED + c];
    s[c] = x; __syncthreads();
    for (int st = 32; st > 0; st >>= 1) { if (c < st) s[c] += s[c + st]; __syncthreads(); }
    float mu = s[0] * (1.0f / EMBED); __syncthreads();
    float d = x - mu;
    s[c] = d * d; __syncthreads();
    for (int st = 32; st > 0; st >>= 1) { if (c < st) s[c] += s[c + st]; __syncthreads(); }
    float var = s[0] * (1.0f / EMBED);
    h[bt * EMBED + c] += d * rsqrtf(var + EPS) * g[c] + bb[c];
}

// ---------------------------------------------------------------- LN2 -> bf16 z
__global__ void k_ln2(const float* __restrict__ h, const float* __restrict__ g,
                      const float* __restrict__ bb, bf16_t* __restrict__ zb) {
    __shared__ float s[EMBED];
    int bt = blockIdx.x, c = threadIdx.x;
    float x = h[bt * EMBED + c];
    s[c] = x; __syncthreads();
    for (int st = 32; st > 0; st >>= 1) { if (c < st) s[c] += s[c + st]; __syncthreads(); }
    float mu = s[0] * (1.0f / EMBED); __syncthreads();
    float d = x - mu;
    s[c] = d * d; __syncthreads();
    for (int st = 32; st > 0; st >>= 1) { if (c < st) s[c] += s[c + st]; __syncthreads(); }
    float var = s[0] * (1.0f / EMBED);
    zb[bt * EMBED + c] = (bf16_t)(d * rsqrtf(var + EPS) * g[c] + bb[c]);
}

// ---------------------------------------------------------------- WMMA bf16 GEMM
// C[M,N] = A[M,K](row-major bf16) @ Bp (pre-packed fragments) + bias (+relu) (+resid).
// 4 waves/block; each wave owns one 16(M) x 64(N) tile.
// Inner loop: A frag = 2 x b128 loads, B frags = 4 x (2 x b128) contiguous loads,
// then 4 WMMAs. Epilogue bounces the 16x64 f32 tile through wave-private LDS so
// global writes become b128 stores (and resid/bf16 side-output are vectorized).
__global__ void k_wmma_gemm(const bf16_t* __restrict__ A, const bf16_t* __restrict__ Bp,
                            const float* __restrict__ bias,
                            float* __restrict__ outF, bf16_t* __restrict__ outB,
                            const float* __restrict__ resid,
                            int M, int N, int K, int relu) {
    __shared__ float ctile[4][16 * 64];             // 4 KB per wave
    int wave = threadIdx.x >> 5;
    int lane = threadIdx.x & 31;
    int tilesM = M >> 4;
    int total  = tilesM * (N >> 6);
    int tile   = blockIdx.x * 4 + wave;
    if (tile >= total) tile = total - 1;            // duplicate work; keeps barrier uniform
    int tm = tile % tilesM, tn = tile / tilesM;
    int m0 = tm << 4, n0 = tn << 6;

    v8f acc[4];
#pragma unroll
    for (int j = 0; j < 4; ++j) acc[j] = (v8f){0.f,0.f,0.f,0.f,0.f,0.f,0.f,0.f};

    const int ktiles = K >> 5;
    // hoisted per-wave base pointers (constant strides inside the loop)
    const bf16_t* ap = A + (size_t)(m0 + (lane & 15)) * K + ((lane >> 4) << 3);
    const bf16_t* bbase = Bp + ((size_t)(n0 >> 4) * ktiles * 512) + (size_t)lane * 16;
    const size_t bstepj = (size_t)ktiles * 512;     // next N-subtile
    for (int kt = 0; kt < ktiles; ++kt) {
        v8bf alo = *(const v8bf*)(ap);              // global_load_b128
        v8bf ahi = *(const v8bf*)(ap + 16);         // global_load_b128
        v16bf a = __builtin_shufflevector(alo, ahi,
                    0,1,2,3,4,5,6,7,8,9,10,11,12,13,14,15);
        v16bf b0 = *(const v16bf*)(bbase);
        v16bf b1 = *(const v16bf*)(bbase + bstepj);
        v16bf b2 = *(const v16bf*)(bbase + 2 * bstepj);
        v16bf b3 = *(const v16bf*)(bbase + 3 * bstepj);
        acc[0] = __builtin_amdgcn_wmma_f32_16x16x32_bf16(false, a, false, b0, (short)0, acc[0], false, false);
        acc[1] = __builtin_amdgcn_wmma_f32_16x16x32_bf16(false, a, false, b1, (short)0, acc[1], false, false);
        acc[2] = __builtin_amdgcn_wmma_f32_16x16x32_bf16(false, a, false, b2, (short)0, acc[2], false, false);
        acc[3] = __builtin_amdgcn_wmma_f32_16x16x32_bf16(false, a, false, b3, (short)0, acc[3], false, false);
        ap += 32;
        bbase += 512;                               // next K-tile
    }

    // ---- epilogue phase 1: accumulators (+bias, relu) -> wave-private LDS tile
    float* my = ctile[wave];
    int rbase = (lane >> 4) << 3;                   // 0 or 8
    int nlo = lane & 15;
#pragma unroll
    for (int j = 0; j < 4; ++j) {
        int cl = j * 16 + nlo;
        float bsv = bias ? bias[n0 + cl] : 0.f;
#pragma unroll
        for (int i = 0; i < 8; ++i) {
            float val = acc[j][i] + bsv;
            if (relu) val = fmaxf(val, 0.f);
            my[(rbase + i) * 64 + cl] = val;
        }
    }
    __syncthreads();
    // ---- epilogue phase 2: read back row-major float4, vectorized global stores
#pragma unroll
    for (int u = 0; u < 8; ++u) {
        int fid = u * 32 + lane;                    // 0..255 float4 slots of 16x64 tile
        int row = fid >> 4;
        int c4  = fid & 15;
        float4 vv = *(const float4*)&my[row * 64 + c4 * 4];
        size_t gidx = (size_t)(m0 + row) * N + n0 + c4 * 4;
        if (resid) {
            float4 rr = *(const float4*)&resid[gidx];
            vv.x += rr.x; vv.y += rr.y; vv.z += rr.z; vv.w += rr.w;
        }
        if (outF) *(float4*)&outF[gidx] = vv;       // global_store_b128
        if (outB) {
            v4bf bb;
            bb[0] = (bf16_t)vv.x; bb[1] = (bf16_t)vv.y;
            bb[2] = (bf16_t)vv.z; bb[3] = (bf16_t)vv.w;
            *(v4bf*)&outB[gidx] = bb;               // global_store_b64
        }
    }
}

// ---------------------------------------------------------------- loss
__global__ void k_loss_init(float* loss) { if (threadIdx.x == 0 && blockIdx.x == 0) *loss = 0.f; }

__global__ void k_loss(const float* __restrict__ logits, const int* __restrict__ targets,
                       float* __restrict__ loss) {
    int row = blockIdx.x;
    int tid = threadIdx.x;
    const float*  lr  = logits + (size_t)row * VOCAB;
    const float4* lr4 = (const float4*)lr;          // VOCAB % 4 == 0, rows 16B-aligned
    __shared__ float sm[256];
    float m = -INFINITY;
    for (int c = tid; c < VOCAB / 4; c += 256) {
        float4 t = lr4[c];
        m = fmaxf(m, fmaxf(fmaxf(t.x, t.y), fmaxf(t.z, t.w)));
    }
    sm[tid] = m; __syncthreads();
    for (int st = 128; st > 0; st >>= 1) {
        if (tid < st) sm[tid] = fmaxf(sm[tid], sm[tid + st]);
        __syncthreads();
    }
    float mx = sm[0]; __syncthreads();
    float s = 0.f;
    for (int c = tid; c < VOCAB / 4; c += 256) {
        float4 t = lr4[c];
        s += __expf(t.x - mx) + __expf(t.y - mx) + __expf(t.z - mx) + __expf(t.w - mx);
    }
    sm[tid] = s; __syncthreads();
    for (int st = 128; st > 0; st >>= 1) {
        if (tid < st) sm[tid] += sm[tid + st];
        __syncthreads();
    }
    if (tid == 0) {
        float lse = __logf(sm[0]) + mx;
        float nll = lse - lr[targets[row]];
        atomicAdd(loss, nll * (1.0f / (float)BT));
    }
}

// ================================================================ launch
extern "C" void kernel_launch(void* const* d_in, const int* in_sizes, int n_in,
                              void* d_out, int out_size, void* d_ws, size_t ws_size,
                              hipStream_t stream) {
    (void)in_sizes; (void)n_in; (void)out_size; (void)ws_size;
    const int*   x    = (const int*)  d_in[0];
    const int*   tgt  = (const int*)  d_in[1];
    const float* tok  = (const float*)d_in[2];
    const float* pos  = (const float*)d_in[3];
    const float* Wq   = (const float*)d_in[4];
    const float* bq   = (const float*)d_in[5];
    const float* Wk   = (const float*)d_in[6];
    const float* bk   = (const float*)d_in[7];
    const float* Wv   = (const float*)d_in[8];
    const float* bv   = (const float*)d_in[9];
    const float* ln1g = (const float*)d_in[10];
    const float* ln1b = (const float*)d_in[11];
    const float* ln2g = (const float*)d_in[12];
    const float* ln2b = (const float*)d_in[13];
    const float* W1   = (const float*)d_in[14];
    const float* b1   = (const float*)d_in[15];
    const float* W2   = (const float*)d_in[16];
    const float* b2   = (const float*)d_in[17];
    const float* Wout = (const float*)d_in[18];
    const float* bout = (const float*)d_in[19];

    float* logits = (float*)d_out;
    float* loss   = logits + (size_t)BT * VOCAB;

    // workspace carve-out (~21 MB)
    char* wp = (char*)d_ws;
    auto carve = [&](size_t bytes) { char* p = wp; wp += (bytes + 255) & ~(size_t)255; return p; };
    float*  h     = (float*) carve((size_t)BT * EMBED * 4);
    float*  attn  = (float*) carve((size_t)BT * EMBED * 4);
    float*  qb    = (float*) carve((size_t)BT * EMBED * 4);
    float*  kb    = (float*) carve((size_t)BT * EMBED * 4);
    float*  vb    = (float*) carve((size_t)BT * EMBED * 4);
    bf16_t* zb    = (bf16_t*)carve((size_t)BT * EMBED * 2);
    bf16_t* act1b = (bf16_t*)carve((size_t)BT * FF * 2);
    bf16_t* hb    = (bf16_t*)carve((size_t)BT * EMBED * 2);
    bf16_t* W1p   = (bf16_t*)carve((size_t)LAYERS * EMBED * FF * 2);
    bf16_t* W2p   = (bf16_t*)carve((size_t)LAYERS * FF * EMBED * 2);
    bf16_t* Woutp = (bf16_t*)carve((size_t)EMBED * VOCAB * 2);

    // pack weights into WMMA B-fragment layout (bf16). Packed Wout = 4 MB -> L2-resident.
    int n;
    for (int l = 0; l < LAYERS; ++l) {
        n = EMBED * FF;
        k_pack_b<<<(n + 255) / 256, 256, 0, stream>>>(
            W1 + (size_t)l * n, W1p + (size_t)l * n, EMBED, FF);
        n = FF * EMBED;
        k_pack_b<<<(n + 255) / 256, 256, 0, stream>>>(
            W2 + (size_t)l * n, W2p + (size_t)l * n, FF, EMBED);
    }
    n = EMBED * VOCAB;
    k_pack_b<<<(n + 255) / 256, 256, 0, stream>>>(Wout, Woutp, EMBED, VOCAB);

    k_embed<<<(BT * EMBED + 255) / 256, 256, 0, stream>>>(x, tok, pos, h);

    const int WHD = HEADS * EMBED * DHEAD;   // per-layer W stride
    const int BHD = HEADS * DHEAD;
    for (int l = 0; l < LAYERS; ++l) {
        k_qkv<<<BT, 192, 0, stream>>>(h,
            Wq + (size_t)l * WHD, bq + (size_t)l * BHD,
            Wk + (size_t)l * WHD, bk + (size_t)l * BHD,
            Wv + (size_t)l * WHD, bv + (size_t)l * BHD,
            qb, kb, vb);
        k_attn<<<BATCH * HEADS * SEQ, 256, 0, stream>>>(qb, kb, vb, attn);
        k_ln1_res<<<BT, EMBED, 0, stream>>>(attn, ln1g + l * EMBED, ln1b + l * EMBED, h);
        k_ln2<<<BT, EMBED, 0, stream>>>(h, ln2g + l * EMBED, ln2b + l * EMBED, zb);
        // FFN1: act1 = relu(z @ W1 + b1)   [8192,64]x[64,256]
        int t1 = (BT / 16) * (FF / 64);
        k_wmma_gemm<<<(t1 + 3) / 4, 128, 0, stream>>>(
            zb, W1p + (size_t)l * EMBED * FF, b1 + (size_t)l * FF,
            nullptr, act1b, nullptr, BT, FF, EMBED, 1);
        // FFN2: h = h + act1 @ W2 + b2     [8192,256]x[256,64]
        int t2 = (BT / 16) * (EMBED / 64);
        k_wmma_gemm<<<(t2 + 3) / 4, 128, 0, stream>>>(
            act1b, W2p + (size_t)l * FF * EMBED, b2 + (size_t)l * EMBED,
            h, nullptr, h, BT, EMBED, FF, 0);
    }

    // logits = h @ Wout + bout             [8192,64]x[64,32000] -> 1.05 GB store
    n = BT * EMBED; k_f2b<<<(n + 255) / 256, 256, 0, stream>>>(h, hb, n);
    int tO = (BT / 16) * (VOCAB / 64);       // 256000 wave-tiles
    k_wmma_gemm<<<(tO + 3) / 4, 128, 0, stream>>>(
        hb, Woutp, bout, logits, nullptr, nullptr, BT, VOCAB, EMBED, 0);

    k_loss_init<<<1, 1, 0, stream>>>(loss);
    k_loss<<<BT, 256, 0, stream>>>(logits, tgt, loss);
}
